// EnhancedFractalTokenProcessor_43112881717570
// MI455X (gfx1250) — compile-verified, matrix operand order checked by reading, not run on hardware
//
#include <hip/hip_runtime.h>
#include <hip/hip_bf16.h>
#include <math.h>

typedef __attribute__((ext_vector_type(16))) __bf16 v16bf;
typedef __attribute__((ext_vector_type(8)))  float  v8f;
typedef unsigned short ushort_t;

#define NTOK 4096
#define DIN  768
#define DMID 512
#define DD   256
#define D4   64
#define DH   128
#define NEXP 51

__device__ __forceinline__ ushort_t f2bf(float f) {
    union { float f; unsigned u; } x; x.f = f;
    unsigned r = x.u + 0x7fffu + ((x.u >> 16) & 1u);
    return (ushort_t)(r >> 16);
}

__device__ __forceinline__ float gelu_exact(float x) {
    return 0.5f * x * (1.0f + erff(x * 0.70710678118654752f));
}

// A fragment (16x32 bf16, row-major source, ld = lda halves)
__device__ __forceinline__ v16bf load_a_frag(const ushort_t* A, int lda, int kBase, int lane) {
    int row = lane & 15;
    int k0  = kBase + ((lane >> 4) << 3);
    union { v16bf v; uint4 q[2]; } u;
    const ushort_t* p = A + (long)row * lda + k0;
    u.q[0] = *(const uint4*)(p);
    u.q[1] = *(const uint4*)(p + 16);
    return u.v;
}

// B fragment from packed layout: [(kt*numNT+nt)*32 + lane]*16 contiguous halves
__device__ __forceinline__ v16bf load_b_frag(const ushort_t* Bp, int kt, int numNT, int nt, int lane) {
    const ushort_t* p = Bp + ((((long)kt * numNT + nt) * 32 + lane) << 4);
    union { v16bf v; uint4 q[2]; } u;
    u.q[0] = *(const uint4*)(p);
    u.q[1] = *(const uint4*)(p + 8);
    return u.v;
}

__device__ __forceinline__ v8f wmma_bf16(v16bf a, v16bf b, v8f c) {
    return __builtin_amdgcn_wmma_f32_16x16x32_bf16(false, a, false, b, (short)0, c, false, false);
}

// ---------------- constant feature branch (spatial + level MLPs) ----------------
__global__ void k_const_feats(const float* spw1, const float* spb1, const float* spw2, const float* spb2,
                              const float* lw1, const float* lb1, const float* lw2, const float* lb2,
                              const int* levels, const int* ph, const int* pw, float* cfeat) {
    __shared__ float hs[64], hl[64];
    int t = threadIdx.x;
    float fph = (float)ph[0], fpw = (float)pw[0], flv = (float)levels[0];
    hs[t] = fmaxf(0.f, fph * spw1[t] + fpw * spw1[64 + t] + spb1[t]);
    hl[t] = fmaxf(0.f, flv * lw1[t] + lb1[t]);
    __syncthreads();
    float os = spb2[t], ol = lb2[t];
    for (int k = 0; k < 64; ++k) {
        os += hs[k] * spw2[k * 64 + t];
        ol += hl[k] * lw2[k * 64 + t];
    }
    cfeat[t]      = os;
    cfeat[64 + t] = ol;
}

// -------- per-token: LN (bf16 out), stats/edge MLPs, enh_in assembly, idx clip --------
__global__ void k_token_stats(const float* __restrict__ tokens, const float* ln_g, const float* ln_b,
                              const float* sw1, const float* sb1, const float* sw2, const float* sb2,
                              const float* ew1, const float* eb1, const float* ew2, const float* eb2,
                              const float* cfeat, const int* levels,
                              ushort_t* __restrict__ xln, ushort_t* __restrict__ enh, int* idxArr) {
    __shared__ float r1[256], r2[256], r3[256];
    __shared__ float h1[64], h2[64];
    __shared__ float scal[4];
    int t = threadIdx.x, n = blockIdx.x;
    const float* x = tokens + (long)n * DIN;
    float s = 0.f, sq = 0.f, ed = 0.f;
    for (int i = t; i < DIN; i += 256) {
        float v = x[i];
        s += v; sq += v * v;
        if (i < DIN - 1) ed += fabsf(x[i + 1] - v);
    }
    r1[t] = s; r2[t] = sq; r3[t] = ed;
    __syncthreads();
    for (int off = 128; off > 0; off >>= 1) {
        if (t < off) { r1[t] += r1[t + off]; r2[t] += r2[t + off]; r3[t] += r3[t + off]; }
        __syncthreads();
    }
    if (t == 0) {
        float mean = r1[0] / (float)DIN;
        float varb = r2[0] / (float)DIN - mean * mean;      // biased (for LN)
        scal[0] = mean;
        scal[1] = rsqrtf(varb + 1e-5f);
        scal[2] = (r2[0] - (float)DIN * mean * mean) / (float)(DIN - 1);  // unbiased var
        scal[3] = r3[0] / (float)(DIN - 1);                  // edge mean over 767 diffs
    }
    __syncthreads();
    float mean = scal[0], rsig = scal[1], varu = scal[2], edge = scal[3];
    for (int i = t; i < DIN; i += 256) {
        float v = (x[i] - mean) * rsig * ln_g[i] + ln_b[i];
        xln[(long)n * DIN + i] = f2bf(v);
    }
    if (t < 64) {
        h1[t] = fmaxf(0.f, varu * sw1[t] + mean * sw1[64 + t] + sb1[t]);
        h2[t] = fmaxf(0.f, edge * ew1[t] + eb1[t]);
    }
    __syncthreads();
    if (t < 64) {
        float o1 = sb2[t], o2 = eb2[t];
        for (int k = 0; k < 64; ++k) {
            o1 += h1[k] * sw2[k * 64 + t];
            o2 += h2[k] * ew2[k * 64 + t];
        }
        long base = (long)n * DD;
        enh[base + t]        = f2bf(o1);            // stats_feat
        enh[base + 64 + t]   = f2bf(o2);            // edge_feat
        enh[base + 128 + t]  = f2bf(cfeat[t]);      // spatial_feat (const)
        enh[base + 192 + t]  = f2bf(cfeat[64 + t]); // level_feat (const)
    }
    if (t == 0) {
        int l = levels[n * 2];
        idxArr[n] = min(max(l, 0), NEXP - 1);
    }
}

// ---------------- weight pack: f32 [batch,K,N] row-major -> bf16 fragment layout ----------------
__global__ void k_pack(const float* __restrict__ src, ushort_t* __restrict__ dst,
                       int K, int N, int batch) {
    long id = (long)blockIdx.x * 256 + threadIdx.x;
    long per = (long)K * N;
    if (id >= per * batch) return;
    int b = (int)(id / per);
    long r = id - (long)b * per;
    int e    = (int)(r & 15);
    int lane = (int)((r >> 4) & 31);
    long tle = r >> 9;
    int numNT = N >> 4;
    int nt = (int)(tle % numNT);
    int kt = (int)(tle / numNT);
    int k = kt * 32 + ((lane >> 4) << 3) + (e & 7) + ((e >> 3) << 4);
    int n = nt * 16 + (lane & 15);
    dst[id] = f2bf(src[(long)b * per + (long)k * N + n]);
}

// ------------- GEMM1: xln[4096,768] @ pw1 -> GELU -> t1 bf16 [4096,512] -------------
// 32 rows x 256 cols per block; B fragments reused across two A row-tiles.
__global__ void k_gemm1(const ushort_t* __restrict__ A, const ushort_t* __restrict__ Bp,
                        const float* __restrict__ bias, ushort_t* __restrict__ out) {
    int lane = threadIdx.x & 31, wave = threadIdx.x >> 5;
    int rowBase = blockIdx.x * 32;
    int colBase = blockIdx.y * 256 + wave * 32;
    const int numNT = DMID >> 4;
    const int KT = DIN / 32;
    int nt0 = colBase >> 4, nt1 = nt0 + 1;
    const ushort_t* Arow0 = A + (long)rowBase * DIN;
    const ushort_t* Arow1 = Arow0 + 16 * DIN;
    v8f c00 = {}, c01 = {}, c10 = {}, c11 = {};
    v16bf a0 = load_a_frag(Arow0, DIN, 0, lane);
    v16bf a1 = load_a_frag(Arow1, DIN, 0, lane);
    v16bf b0 = load_b_frag(Bp, 0, numNT, nt0, lane);
    v16bf b1 = load_b_frag(Bp, 0, numNT, nt1, lane);
    #pragma unroll 4
    for (int kt = 0; kt < KT; ++kt) {
        v16bf na0 = a0, na1 = a1, nb0 = b0, nb1 = b1;
        if (kt + 1 < KT) {      // prefetch next fragments into registers
            na0 = load_a_frag(Arow0, DIN, (kt + 1) * 32, lane);
            na1 = load_a_frag(Arow1, DIN, (kt + 1) * 32, lane);
            nb0 = load_b_frag(Bp, kt + 1, numNT, nt0, lane);
            nb1 = load_b_frag(Bp, kt + 1, numNT, nt1, lane);
        }
        if (kt + 2 < KT) {      // L2/L0 prefetch of the B stream (global_prefetch_b8)
            __builtin_prefetch(Bp + ((((long)(kt + 2) * numNT + nt0) * 32 + lane) << 4), 0, 3);
            __builtin_prefetch(Bp + ((((long)(kt + 2) * numNT + nt1) * 32 + lane) << 4), 0, 3);
        }
        c00 = wmma_bf16(a0, b0, c00);
        c01 = wmma_bf16(a0, b1, c01);
        c10 = wmma_bf16(a1, b0, c10);
        c11 = wmma_bf16(a1, b1, c11);
        a0 = na0; a1 = na1; b0 = nb0; b1 = nb1;
    }
    int r0 = (lane >> 4) << 3, cIn = lane & 15;
    for (int v = 0; v < 8; ++v) {
        int rowA = rowBase + r0 + v;
        int rowB = rowA + 16;
        int ca = colBase + cIn, cb = colBase + 16 + cIn;
        float ba = bias[ca], bb = bias[cb];
        out[(long)rowA * DMID + ca] = f2bf(gelu_exact(c00[v] + ba));
        out[(long)rowA * DMID + cb] = f2bf(gelu_exact(c01[v] + bb));
        out[(long)rowB * DMID + ca] = f2bf(gelu_exact(c10[v] + ba));
        out[(long)rowB * DMID + cb] = f2bf(gelu_exact(c11[v] + bb));
    }
}

// ------------- GEMM2: t1 @ pw2 + pb2 + emb[idx] -> h f32 [4096,256] -------------
__global__ void k_gemm2(const ushort_t* __restrict__ A, const ushort_t* __restrict__ Bp,
                        const float* __restrict__ bias, const float* __restrict__ emb,
                        const int* __restrict__ idxArr, float* __restrict__ h) {
    int lane = threadIdx.x & 31, wave = threadIdx.x >> 5;
    int rowBase = blockIdx.x * 32;
    int colBase = wave * 32;
    const int numNT = DD >> 4;
    const int KT = DMID / 32;
    int nt0 = colBase >> 4, nt1 = nt0 + 1;
    const ushort_t* Arow0 = A + (long)rowBase * DMID;
    const ushort_t* Arow1 = Arow0 + 16 * DMID;
    v8f c00 = {}, c01 = {}, c10 = {}, c11 = {};
    v16bf a0 = load_a_frag(Arow0, DMID, 0, lane);
    v16bf a1 = load_a_frag(Arow1, DMID, 0, lane);
    v16bf b0 = load_b_frag(Bp, 0, numNT, nt0, lane);
    v16bf b1 = load_b_frag(Bp, 0, numNT, nt1, lane);
    #pragma unroll 4
    for (int kt = 0; kt < KT; ++kt) {
        v16bf na0 = a0, na1 = a1, nb0 = b0, nb1 = b1;
        if (kt + 1 < KT) {
            na0 = load_a_frag(Arow0, DMID, (kt + 1) * 32, lane);
            na1 = load_a_frag(Arow1, DMID, (kt + 1) * 32, lane);
            nb0 = load_b_frag(Bp, kt + 1, numNT, nt0, lane);
            nb1 = load_b_frag(Bp, kt + 1, numNT, nt1, lane);
        }
        if (kt + 2 < KT) {
            __builtin_prefetch(Bp + ((((long)(kt + 2) * numNT + nt0) * 32 + lane) << 4), 0, 3);
            __builtin_prefetch(Bp + ((((long)(kt + 2) * numNT + nt1) * 32 + lane) << 4), 0, 3);
        }
        c00 = wmma_bf16(a0, b0, c00);
        c01 = wmma_bf16(a0, b1, c01);
        c10 = wmma_bf16(a1, b0, c10);
        c11 = wmma_bf16(a1, b1, c11);
        a0 = na0; a1 = na1; b0 = nb0; b1 = nb1;
    }
    int r0 = (lane >> 4) << 3, cIn = lane & 15;
    for (int v = 0; v < 8; ++v) {
        int rowA = rowBase + r0 + v;
        int rowB = rowA + 16;
        int idA = idxArr[rowA], idB = idxArr[rowB];
        int ca = colBase + cIn, cb = colBase + 16 + cIn;
        h[(long)rowA * DD + ca] = c00[v] + bias[ca] + emb[(long)idA * DD + ca];
        h[(long)rowA * DD + cb] = c01[v] + bias[cb] + emb[(long)idA * DD + cb];
        h[(long)rowB * DD + ca] = c10[v] + bias[ca] + emb[(long)idB * DD + ca];
        h[(long)rowB * DD + cb] = c11[v] + bias[cb] + emb[(long)idB * DD + cb];
    }
}

// ------- GEMM3: enh_in @ fw + fb -> LN(fg,fbeta) -> GELU; h += 0.3*enh; hb = bf16(h) -------
__global__ void k_gemm3(const ushort_t* __restrict__ A, const ushort_t* __restrict__ Bp,
                        const float* __restrict__ fb, const float* __restrict__ fg,
                        const float* __restrict__ fbeta,
                        float* __restrict__ h, ushort_t* __restrict__ hb) {
    __shared__ float tile[16 * DD];
    __shared__ float mr[16], rr[16];
    int t = threadIdx.x;
    int lane = t & 31, wave = t >> 5;
    int rowBase = blockIdx.x * 16;
    int colBase = wave * 32;
    const int numNT = DD >> 4;
    const int KT = DD / 32;
    int nt0 = colBase >> 4, nt1 = nt0 + 1;
    const ushort_t* Arow = A + (long)rowBase * DD;
    v8f c0 = {}, c1 = {};
    v16bf a = load_a_frag(Arow, DD, 0, lane);
    v16bf b0 = load_b_frag(Bp, 0, numNT, nt0, lane);
    v16bf b1 = load_b_frag(Bp, 0, numNT, nt1, lane);
    #pragma unroll
    for (int kt = 0; kt < KT; ++kt) {
        v16bf na = a, nb0 = b0, nb1 = b1;
        if (kt + 1 < KT) {
            na  = load_a_frag(Arow, DD, (kt + 1) * 32, lane);
            nb0 = load_b_frag(Bp, kt + 1, numNT, nt0, lane);
            nb1 = load_b_frag(Bp, kt + 1, numNT, nt1, lane);
        }
        c0 = wmma_bf16(a, b0, c0);
        c1 = wmma_bf16(a, b1, c1);
        a = na; b0 = nb0; b1 = nb1;
    }
    int r0 = (lane >> 4) << 3, cIn = lane & 15;
    for (int v = 0; v < 8; ++v) {
        int ca = colBase + cIn, cb = colBase + 16 + cIn;
        tile[(r0 + v) * DD + ca] = c0[v] + fb[ca];
        tile[(r0 + v) * DD + cb] = c1[v] + fb[cb];
    }
    __syncthreads();
    if (t < 16) {
        const float* rw = &tile[t * DD];
        float s = 0.f, sq = 0.f;
        for (int i = 0; i < DD; ++i) { float v = rw[i]; s += v; sq += v * v; }
        float m = s / (float)DD;
        mr[t] = m;
        rr[t] = rsqrtf(sq / (float)DD - m * m + 1e-5f);
    }
    __syncthreads();
    {
        int rowL = t >> 4, cb0 = (t & 15) << 4;
        int gRow = rowBase + rowL;
        float m = mr[rowL], rs = rr[rowL];
        for (int j = 0; j < 16; ++j) {
            int col = cb0 + j;
            float e = (tile[rowL * DD + col] - m) * rs * fg[col] + fbeta[col];
            float g = gelu_exact(e);
            long o = (long)gRow * DD + col;
            float hv = h[o] + 0.3f * g;
            h[o]  = hv;
            hb[o] = f2bf(hv);
        }
    }
}

// ---------------- expert bucketing ----------------
__global__ void k_zero(int* counters) { if (threadIdx.x < 64) counters[threadIdx.x] = 0; }

__global__ void k_count(const int* idxArr, int* counters) {
    int n = blockIdx.x * 256 + threadIdx.x;
    if (n < NTOK) atomicAdd(&counters[idxArr[n]], 1);
}

__global__ void k_scan(const int* counters, int* offsets, int* cursor) {
    if (threadIdx.x == 0) {
        int s = 0;
        for (int e = 0; e < NEXP; ++e) { offsets[e] = s; cursor[e] = s; s += counters[e]; }
        offsets[NEXP] = s;
    }
}

__global__ void k_scatter(const int* idxArr, int* cursor, int* perm) {
    int n = blockIdx.x * 256 + threadIdx.x;
    if (n < NTOK) {
        int p = atomicAdd(&cursor[idxArr[n]], 1);
        perm[p] = n;
    }
}

// ---- MoE: per-expert tiles, gather h rows to LDS, WMMA vs packed aW[e], LN+GELU -> y f32 ----
__global__ void k_moe(const ushort_t* __restrict__ hb, const ushort_t* __restrict__ awp,
                      const float* __restrict__ ab, const float* __restrict__ ag,
                      const float* __restrict__ abt,
                      const int* __restrict__ offsets, const int* __restrict__ perm,
                      float* __restrict__ y) {
    int e = blockIdx.x;
    int tileStart = blockIdx.y * 16;
    int beg = offsets[e];
    int cnt = offsets[e + 1] - beg;
    if (tileStart >= cnt) return;

    __shared__ int toks[16];
    __shared__ __align__(16) ushort_t aT[16 * DD];
    __shared__ float tile[16 * DD];
    __shared__ float mr[16], rr[16];

    int t = threadIdx.x;
    if (t < 16) {
        int r = tileStart + t;
        toks[t] = (r < cnt) ? perm[beg + r] : perm[beg];
    }
    __syncthreads();
    {   // stage 16x256 bf16 A tile into LDS (gathered rows)
        int rowL = t >> 4, cb0 = (t & 15) << 4;
        const uint4* src = (const uint4*)(hb + (long)toks[rowL] * DD + cb0);
        uint4* dst = (uint4*)(aT + rowL * DD + cb0);
        dst[0] = src[0];
        dst[1] = src[1];
    }
    __syncthreads();

    int lane = t & 31, wave = t >> 5;
    int colBase = wave * 32;
    const int numNT = DD >> 4;
    const int KT = DD / 32;
    int nt0 = colBase >> 4, nt1 = nt0 + 1;
    const ushort_t* Bp = awp + (long)e * DD * DD;
    v8f c0 = {}, c1 = {};
    v16bf b0 = load_b_frag(Bp, 0, numNT, nt0, lane);
    v16bf b1 = load_b_frag(Bp, 0, numNT, nt1, lane);
    #pragma unroll
    for (int kt = 0; kt < KT; ++kt) {
        v16bf nb0 = b0, nb1 = b1;
        if (kt + 1 < KT) {   // double-buffer the global (expert weight) stream
            nb0 = load_b_frag(Bp, kt + 1, numNT, nt0, lane);
            nb1 = load_b_frag(Bp, kt + 1, numNT, nt1, lane);
        }
        if (kt + 2 < KT) {
            __builtin_prefetch(Bp + ((((long)(kt + 2) * numNT + nt0) * 32 + lane) << 4), 0, 3);
            __builtin_prefetch(Bp + ((((long)(kt + 2) * numNT + nt1) * 32 + lane) << 4), 0, 3);
        }
        v16bf a = load_a_frag((const ushort_t*)aT, DD, kt * 32, lane);  // LDS, low latency
        c0 = wmma_bf16(a, b0, c0);
        c1 = wmma_bf16(a, b1, c1);
        b0 = nb0; b1 = nb1;
    }
    int r0 = (lane >> 4) << 3, cIn = lane & 15;
    for (int v = 0; v < 8; ++v) {
        int ca = colBase + cIn, cb = colBase + 16 + cIn;
        tile[(r0 + v) * DD + ca] = c0[v] + ab[(long)e * DD + ca];
        tile[(r0 + v) * DD + cb] = c1[v] + ab[(long)e * DD + cb];
    }
    __syncthreads();
    if (t < 16) {
        const float* rw = &tile[t * DD];
        float s = 0.f, sq = 0.f;
        for (int i = 0; i < DD; ++i) { float v = rw[i]; s += v; sq += v * v; }
        float m = s / (float)DD;
        mr[t] = m;
        rr[t] = rsqrtf(sq / (float)DD - m * m + 1e-5f);
    }
    __syncthreads();
    {
        int rowL = t >> 4, cb0 = (t & 15) << 4;
        int rIdx = tileStart + rowL;
        if (rIdx < cnt) {
            int tok = toks[rowL];
            float m = mr[rowL], rs = rr[rowL];
            for (int j = 0; j < 16; ++j) {
                int col = cb0 + j;
                float vv = (tile[rowL * DD + col] - m) * rs * ag[(long)e * DD + col] + abt[(long)e * DD + col];
                y[(long)tok * DD + col] = gelu_exact(vv);
            }
        }
    }
}

// ---------------- gate: sigmoid MLP, scale, write final output ----------------
__global__ void k_gate(const float* __restrict__ y, const float* __restrict__ dw1,
                       const float* __restrict__ db1, const float* __restrict__ dw2,
                       const float* __restrict__ db2, float* __restrict__ out) {
    __shared__ float ys[DD];
    __shared__ float red[DH];
    int n = blockIdx.x, t = threadIdx.x; // 128 threads
    ys[t]       = y[(long)n * DD + t];
    ys[t + DH]  = y[(long)n * DD + t + DH];
    __syncthreads();
    float acc = db1[t];
    for (int d = 0; d < DD; ++d) acc += ys[d] * dw1[d * DH + t];
    red[t] = fmaxf(acc, 0.f) * dw2[t];
    __syncthreads();
    for (int off = DH / 2; off > 0; off >>= 1) {
        if (t < off) red[t] += red[t + off];
        __syncthreads();
    }
    float w = 1.f / (1.f + expf(-(red[0] + db2[0])));
    float sc = 0.8f + 0.4f * w;
    out[(long)n * DD + t]      = ys[t] * sc;
    out[(long)n * DD + t + DH] = ys[t + DH] * sc;
}

extern "C" void kernel_launch(void* const* d_in, const int* in_sizes, int n_in,
                              void* d_out, int out_size, void* d_ws, size_t ws_size,
                              hipStream_t stream) {
    const float* tokens = (const float*)d_in[0];
    const float* ln_g   = (const float*)d_in[1];
    const float* ln_b   = (const float*)d_in[2];
    const float* pw1    = (const float*)d_in[3];
    const float* pb1    = (const float*)d_in[4];
    const float* pw2    = (const float*)d_in[5];
    const float* pb2    = (const float*)d_in[6];
    const float* emb    = (const float*)d_in[7];
    const float* sw1    = (const float*)d_in[8];
    const float* sb1    = (const float*)d_in[9];
    const float* sw2    = (const float*)d_in[10];
    const float* sb2    = (const float*)d_in[11];
    const float* ew1    = (const float*)d_in[12];
    const float* eb1    = (const float*)d_in[13];
    const float* ew2    = (const float*)d_in[14];
    const float* eb2    = (const float*)d_in[15];
    const float* spw1   = (const float*)d_in[16];
    const float* spb1   = (const float*)d_in[17];
    const float* spw2   = (const float*)d_in[18];
    const float* spb2   = (const float*)d_in[19];
    const float* lw1    = (const float*)d_in[20];
    const float* lb1    = (const float*)d_in[21];
    const float* lw2    = (const float*)d_in[22];
    const float* lb2    = (const float*)d_in[23];
    const float* fw     = (const float*)d_in[24];
    const float* fb     = (const float*)d_in[25];
    const float* fg     = (const float*)d_in[26];
    const float* fbeta  = (const float*)d_in[27];
    const float* aW     = (const float*)d_in[28];
    const float* ab     = (const float*)d_in[29];
    const float* ag     = (const float*)d_in[30];
    const float* abt    = (const float*)d_in[31];
    const float* dw1    = (const float*)d_in[32];
    const float* db1    = (const float*)d_in[33];
    const float* dw2    = (const float*)d_in[34];
    const float* db2    = (const float*)d_in[35];
    const int*   levels = (const int*)d_in[36];
    const int*   ph     = (const int*)d_in[37];
    const int*   pw     = (const int*)d_in[38];
    float* out = (float*)d_out;

    // ---- workspace carving ----
    char* ws = (char*)d_ws;
    size_t off = 0;
    auto al = [&](size_t bytes) { size_t r = off; off += (bytes + 255) & ~(size_t)255; return r; };
    ushort_t* xln  = (ushort_t*)(ws + al((size_t)NTOK * DIN * 2));
    ushort_t* t1   = (ushort_t*)(ws + al((size_t)NTOK * DMID * 2));
    float*    h    = (float*)   (ws + al((size_t)NTOK * DD * 4));
    ushort_t* hb   = (ushort_t*)(ws + al((size_t)NTOK * DD * 2));
    ushort_t* enh  = (ushort_t*)(ws + al((size_t)NTOK * DD * 2));
    float*    yb   = (float*)   (ws + al((size_t)NTOK * DD * 4));
    ushort_t* pw1p = (ushort_t*)(ws + al((size_t)DIN * DMID * 2));
    ushort_t* pw2p = (ushort_t*)(ws + al((size_t)DMID * DD * 2));
    ushort_t* fwp  = (ushort_t*)(ws + al((size_t)DD * DD * 2));
    ushort_t* awp  = (ushort_t*)(ws + al((size_t)NEXP * DD * DD * 2));
    int* idxArr    = (int*)(ws + al((size_t)NTOK * 4));
    int* counters  = (int*)(ws + al(64 * 4));
    int* offsets   = (int*)(ws + al(64 * 4));
    int* cursor    = (int*)(ws + al(64 * 4));
    int* perm      = (int*)(ws + al((size_t)NTOK * 4));
    float* cfeat   = (float*)(ws + al(128 * 4));

    // constant feature MLPs (spatial + level)
    k_const_feats<<<1, 64, 0, stream>>>(spw1, spb1, spw2, spb2, lw1, lb1, lw2, lb2,
                                        levels, ph, pw, cfeat);
    // per-token LN + stats/edge feats + enh assembly + idx
    k_token_stats<<<NTOK, 256, 0, stream>>>(tokens, ln_g, ln_b, sw1, sb1, sw2, sb2,
                                            ew1, eb1, ew2, eb2, cfeat, levels,
                                            xln, enh, idxArr);
    // expert bucketing
    k_zero<<<1, 64, 0, stream>>>(counters);
    k_count<<<NTOK / 256, 256, 0, stream>>>(idxArr, counters);
    k_scan<<<1, 64, 0, stream>>>(counters, offsets, cursor);
    k_scatter<<<NTOK / 256, 256, 0, stream>>>(idxArr, cursor, perm);
    // weight packing (f32 -> bf16 fragment layout)
    k_pack<<<(DIN * DMID) / 256, 256, 0, stream>>>(pw1, pw1p, DIN, DMID, 1);
    k_pack<<<(DMID * DD) / 256, 256, 0, stream>>>(pw2, pw2p, DMID, DD, 1);
    k_pack<<<(DD * DD) / 256, 256, 0, stream>>>(fw, fwp, DD, DD, 1);
    k_pack<<<(NEXP * DD * DD) / 256, 256, 0, stream>>>(aW, awp, DD, DD, NEXP);
    // main GEMM chain (32-row tiles on the two big GEMMs)
    k_gemm1<<<dim3(NTOK / 32, DMID / 256), 256, 0, stream>>>(xln, pw1p, pb1, t1);
    k_gemm2<<<dim3(NTOK / 32, 1), 256, 0, stream>>>(t1, pw2p, pb2, emb, idxArr, h);
    k_gemm3<<<dim3(NTOK / 16, 1), 256, 0, stream>>>(enh, fwp, fb, fg, fbeta, h, hb);
    // MoE expert GEMM + per-expert LN + GELU
    k_moe<<<dim3(NEXP, NTOK / 16), 256, 0, stream>>>(hb, awp, ab, ag, abt, offsets, perm, yb);
    // gate + final scale
    k_gate<<<NTOK, DH, 0, stream>>>(yb, dw1, db1, dw2, db2, out);
}
